// Esp_3736621547678
// MI455X (gfx1250) — matrix operand order repaired and checked by reading, not compiled, not run on hardware
//
#include <hip/hip_runtime.h>
#include <hip/hip_bf16.h>
#include <math.h>

// Problem constants (from reference): B=2,H=8,N=256,L=64, TAU=0.001, T=0.1
#define BH    16          // B*H
#define NN    256
#define LL    64
#define BHL   1024        // BH*LL
#define NEG_INV_TAU (-1000.0f)
#define TEMP  0.1f

typedef __attribute__((ext_vector_type(16))) __bf16 v16bf;
typedef __attribute__((ext_vector_type(8)))  float  v8f;

// ---------------------------------------------------------------------------
// Kernel 1: per (b,h,l,which) column -> sorted values + softmax inv-denoms
// grid 2*BHL blocks of 256 threads
// ---------------------------------------------------------------------------
__global__ __launch_bounds__(256) void prep_kernel(
    const float* __restrict__ X, const float* __restrict__ Y,
    float* sX, float* rX, float* iX,
    float* sY, float* rY, float* iY)
{
    const int which = blockIdx.x & 1;
    const int idx   = blockIdx.x >> 1;        // [0, BHL)
    const int bh    = idx >> 6;
    const int l     = idx & 63;
    const float* src = which ? Y : X;
    float* sA = which ? sY : sX;
    float* rA = which ? rY : rX;
    float* iA = which ? iY : iX;

    __shared__ float ls[NN];
    __shared__ float lr[NN];
    const int t = threadIdx.x;
    const float v = src[((size_t)bh * NN + t) * LL + l];
    ls[t] = v;
    __syncthreads();

    // descending rank (stable via index tie-break)
    int rank = 0;
    for (int m = 0; m < NN; ++m) {
        const float u = ls[m];
        rank += (u > v) || (u == v && m < t);
    }
    lr[rank] = v;          // lr = sorted descending
    __syncthreads();

    // denom_i = sum_j exp(-(s_j - srt_i)^2 / tau)
    const float ri = lr[t];
    float d = 0.0f;
    for (int j = 0; j < NN; ++j) {
        const float df = ls[j] - ri;
        d += __expf(df * df * NEG_INV_TAU);
    }
    const size_t o = (size_t)idx * NN + t;
    sA[o] = v;
    rA[o] = ri;
    iA[o] = 1.0f / d;
}

// ---------------------------------------------------------------------------
// Kernel 2: cost[bh][j][k] = ||X_j - Y_k||_2  (LDS-tiled, 16x16 tile/block)
// grid BH*16*16 blocks of 256 threads
// ---------------------------------------------------------------------------
__global__ __launch_bounds__(256) void cost_kernel(
    const float* __restrict__ X, const float* __restrict__ Y,
    float* __restrict__ cost)
{
    const int bh = blockIdx.x >> 8;
    const int tt = blockIdx.x & 255;
    const int jt = tt >> 4, kt = tt & 15;
    __shared__ float xs[16][65];
    __shared__ float ys[16][65];
    const int tid = threadIdx.x;
    for (int e = tid; e < 16 * 64; e += 256) {
        const int row = e >> 6, col = e & 63;
        xs[row][col] = X[((size_t)bh * NN + jt * 16 + row) * LL + col];
        ys[row][col] = Y[((size_t)bh * NN + kt * 16 + row) * LL + col];
    }
    __syncthreads();
    const int jl = tid >> 4, kl = tid & 15;
    float dot = 0.f, x2 = 0.f, y2 = 0.f;
    for (int c = 0; c < 64; ++c) {
        const float xv = xs[jl][c], yv = ys[kl][c];
        dot = fmaf(xv, yv, dot);
        x2  = fmaf(xv, xv, x2);
        y2  = fmaf(yv, yv, y2);
    }
    const float cc = sqrtf(fmaxf(x2 + y2 - 2.0f * dot, 1e-12f));
    cost[((size_t)bh * NN + jt * 16 + jl) * NN + kt * 16 + kl] = cc;
}

// ---------------------------------------------------------------------------
// Shared WMMA Gamma machinery: one block per (b,h,l); 16 waves (512 thr).
// Wave w owns j-tile jt=w; accumulates all 16 k-tiles (acc[16] of v8f).
// A = Pu^T (16x32 bf16 per chunk, UNNORMALIZED exps), per-lane A layout:
//   M = lane&15 ; K = h + 16*(lane>=16)
// B = Pv * (invd_x[i]*invd_y[i]) (32x16 bf16), staged to LDS in B layout:
//   N = lane&15 ; K = h + 16*(lane>=16)
// Both inverse softmax denominators are folded into B (per-K scale),
// mathematically identical to normalizing each matrix separately.
// ---------------------------------------------------------------------------
template <int MODE>   // 0 = swds pass, 1 = weighted-output pass
__device__ inline void gamma_pass(
    const float* sX, const float* rX, const float* iX,
    const float* sY, const float* rY, const float* iY,
    const float* cost, const float* weights,
    float* swds, float* out)
{
    const int bhl = blockIdx.x;
    const int bh  = bhl >> 6;
    const int l   = bhl & 63;
    const int tid  = threadIdx.x;
    const int wave = tid >> 5;
    const int lane = tid & 31;

    __shared__ float sx[NN], rx[NN], sy[NN], ry[NN], cb[NN];
    __shared__ __align__(16) unsigned short Bf[16][32][16];  // [kt][lane][h]
    __shared__ float red[16];

    const size_t o = (size_t)bhl * NN;
    if (tid < NN) {
        sx[tid] = sX[o + tid];  rx[tid] = rX[o + tid];
        sy[tid] = sY[o + tid];  ry[tid] = rY[o + tid];
        cb[tid] = iX[o + tid] * iY[o + tid];   // combined 1/(dx_i * dy_i)
    }
    __syncthreads();

    const int jt     = wave;
    const int klocal = lane & 15;
    const int hiLane = lane >> 4;              // 0 or 1
    const float sxj  = sx[jt * 16 + klocal];   // s_x[j] for this lane's A row

    v8f acc[16];
    const v8f vzero = {0.f, 0.f, 0.f, 0.f, 0.f, 0.f, 0.f, 0.f};
#pragma unroll
    for (int kt = 0; kt < 16; ++kt) acc[kt] = vzero;

    for (int ic = 0; ic < 8; ++ic) {
        // --- stage B fragments (Pv * comb) to LDS: thread <-> (kt=wave, lane)
        {
            const int kcol  = wave * 16 + klocal;
            const float syk = sy[kcol];
            const int ibase = ic * 32 + hiLane * 16;
            union { uint4 q[2]; __bf16 b[16]; } ub;
#pragma unroll
            for (int h = 0; h < 16; ++h) {
                const float d = syk - ry[ibase + h];
                ub.b[h] = (__bf16)(__expf(d * d * NEG_INV_TAU) * cb[ibase + h]);
            }
            uint4* dst = reinterpret_cast<uint4*>(&Bf[wave][lane][0]);
            dst[0] = ub.q[0]; dst[1] = ub.q[1];
        }
        // --- A fragment (unnormalized Pu^T exps) in registers
        v16bf av;
#pragma unroll
        for (int h = 0; h < 16; ++h) {
            const int i = ic * 32 + hiLane * 16 + h;
            const float d = sxj - rx[i];
            av[h] = (__bf16)__expf(d * d * NEG_INV_TAU);
        }
        __syncthreads();
        // --- 16 WMMAs over this K-chunk
#pragma unroll
        for (int kt = 0; kt < 16; ++kt) {
            union { v16bf v; uint4 q[2]; } ubv;
            const uint4* src = reinterpret_cast<const uint4*>(&Bf[kt][lane][0]);
            ubv.q[0] = src[0]; ubv.q[1] = src[1];
            acc[kt] = __builtin_amdgcn_wmma_f32_16x16x32_bf16(
                false, av, false, ubv.v, (short)0, acc[kt], false, false);
        }
        __syncthreads();
    }

    // C/D layout: element (r, lane) -> M = r + 8*hiLane, N = lane&15
    const int Mbase = jt * 16 + hiLane * 8;

    if (MODE == 0) {
        float partial = 0.f;
        const float* cbh = cost + (size_t)bh * NN * NN;
#pragma unroll
        for (int kt = 0; kt < 16; ++kt) {
            const int k = kt * 16 + klocal;
#pragma unroll
            for (int r = 0; r < 8; ++r)
                partial = fmaf(acc[kt][r], cbh[(Mbase + r) * NN + k], partial);
        }
        for (int off = 16; off > 0; off >>= 1)
            partial += __shfl_down(partial, off, 32);
        if (lane == 0) red[wave] = partial;
        __syncthreads();
        if (tid == 0) {
            float s = 0.f;
            for (int w = 0; w < 16; ++w) s += red[w];
            swds[bhl] = s;
        }
    } else {
        const float w = weights[bh * LL + l];
        float* obh = out + (size_t)bh * NN * NN;
#pragma unroll
        for (int kt = 0; kt < 16; ++kt) {
            const int k = kt * 16 + klocal;
#pragma unroll
            for (int r = 0; r < 8; ++r)
                unsafeAtomicAdd(&obh[(Mbase + r) * NN + k], acc[kt][r] * w);
        }
    }
}

__global__ __launch_bounds__(512) void swds_kernel(
    const float* sX, const float* rX, const float* iX,
    const float* sY, const float* rY, const float* iY,
    const float* cost, float* swds)
{
    gamma_pass<0>(sX, rX, iX, sY, rY, iY, cost, nullptr, swds, nullptr);
}

__global__ __launch_bounds__(512) void out_kernel(
    const float* sX, const float* rX, const float* iX,
    const float* sY, const float* rY, const float* iY,
    const float* weights, float* out)
{
    gamma_pass<1>(sX, rX, iX, sY, rY, iY, nullptr, weights, nullptr, out);
}

// ---------------------------------------------------------------------------
// Kernel 4: weights = softmax(-T * swds) over L, per (b,h)
// ---------------------------------------------------------------------------
__global__ __launch_bounds__(64) void weights_kernel(
    const float* __restrict__ swds, float* __restrict__ weights)
{
    const int bh = blockIdx.x;
    const int t  = threadIdx.x;
    __shared__ float buf[64];
    const float v = swds[bh * LL + t];
    buf[t] = v;
    __syncthreads();
    for (int off = 32; off > 0; off >>= 1) {
        if (t < off) buf[t] = fmaxf(buf[t], buf[t + off]);
        __syncthreads();
    }
    const float mx = buf[0];
    __syncthreads();
    const float e = __expf(-TEMP * (v - mx));
    buf[t] = e;
    __syncthreads();
    for (int off = 32; off > 0; off >>= 1) {
        if (t < off) buf[t] += buf[t + off];
        __syncthreads();
    }
    weights[bh * LL + t] = e / buf[0];
}

__global__ __launch_bounds__(256) void zero_kernel(float* out, int n)
{
    const int i = blockIdx.x * 256 + threadIdx.x;
    if (i < n) out[i] = 0.0f;
}

// ---------------------------------------------------------------------------
extern "C" void kernel_launch(void* const* d_in, const int* in_sizes, int n_in,
                              void* d_out, int out_size, void* d_ws, size_t ws_size,
                              hipStream_t stream)
{
    const float* X = (const float*)d_in[0];
    const float* Y = (const float*)d_in[1];
    float* out = (float*)d_out;
    float* ws  = (float*)d_ws;

    // workspace layout (floats): 6 * BHL*256 + BH*256*256 + 2*BHL  (~10.5 MB)
    float* sX   = ws;
    float* rX   = sX + (size_t)BHL * NN;
    float* iX   = rX + (size_t)BHL * NN;
    float* sY   = iX + (size_t)BHL * NN;
    float* rY   = sY + (size_t)BHL * NN;
    float* iY   = rY + (size_t)BHL * NN;
    float* cost = iY + (size_t)BHL * NN;
    float* swds = cost + (size_t)BH * NN * NN;
    float* wts  = swds + BHL;

    prep_kernel   <<<2 * BHL,      256, 0, stream>>>(X, Y, sX, rX, iX, sY, rY, iY);
    cost_kernel   <<<BH * 16 * 16, 256, 0, stream>>>(X, Y, cost);
    swds_kernel   <<<BHL,          512, 0, stream>>>(sX, rX, iX, sY, rY, iY, cost, swds);
    weights_kernel<<<BH,            64, 0, stream>>>(swds, wts);
    zero_kernel   <<<(BH * NN * NN + 255) / 256, 256, 0, stream>>>(out, BH * NN * NN);
    out_kernel    <<<BHL,          512, 0, stream>>>(sX, rX, iX, sY, rY, iY, wts, out);
}
